// InvariantPointAttention_SURF_BB_SFM_2894807957715
// MI455X (gfx1250) — matrix operand order repaired and checked by reading, not compiled, not run on hardware
//
#include <hip/hip_runtime.h>
#include <hip/hip_bf16.h>

// ======================= CDNA5 WMMA plumbing =======================
typedef __bf16 bf16;
typedef __attribute__((ext_vector_type(16))) __bf16 v16bf;
typedef __attribute__((ext_vector_type(8)))  float  v8f;
typedef __attribute__((ext_vector_type(8)))  unsigned v8u;

#define DEV static __device__ __forceinline__

DEV bf16 f2bf(float f) { return (bf16)f; }   // lowers to v_cvt_pk_bf16_f32
DEV unsigned short bf2bits(bf16 b) { return __builtin_bit_cast(unsigned short, b); }
DEV unsigned pack2bf(float lo, float hi) {
  return (unsigned)bf2bits(f2bf(lo)) | ((unsigned)bf2bits(f2bf(hi)) << 16);
}

// A-matrix K-pair base (ISA 7.12.2, 16-bit A 16x32, interleaved):
// VGPR v<4 : K = 2v + 8*half ; v>=4 : K = 16 + 2(v-4) + 8*half
DEV int koff_a(int v, int half) {
  return (v < 4) ? (2 * v + 8 * half) : (16 + 2 * (v - 4) + 8 * half);
}
// B-matrix K-pair base (same interleaved K order as A):
// VGPR v<4 : K = 2v + 8*half ; v>=4 : K = 16 + 2(v-4) + 8*half
DEV int koff_b(int v, int half) { return (v < 4) ? (2 * v + 8 * half) : (16 + 2 * (v - 4) + 8 * half); }

DEV v8f wmma_bf16(v16bf a, v16bf b, v8f c) {
  return __builtin_amdgcn_wmma_f32_16x16x32_bf16(false, a, false, b, (short)0, c,
                                                 false, false);
}

// ---- UNGUARDED fragment loaders (caller guarantees padded shapes) ----
DEV v16bf afrag_f32(const float* A, long lda, int m0, int k0) {
  int lane = threadIdx.x & 31, half = lane >> 4;
  const float* row = A + (long)(m0 + (lane & 15)) * lda + k0;
  v16bf r;
#pragma unroll
  for (int v = 0; v < 8; ++v) {
    float2 f = *(const float2*)(row + koff_a(v, half));
    r[2 * v] = f2bf(f.x);
    r[2 * v + 1] = f2bf(f.y);
  }
  return r;
}
// B fragment from f32 N-major storage S[n][k]
DEV v16bf bfrag_nmaj(const float* S, long ldb, int n0, int k0) {
  int lane = threadIdx.x & 31, half = lane >> 4;
  const float* row = S + (long)(n0 + (lane & 15)) * ldb + k0;
  v16bf r;
#pragma unroll
  for (int v = 0; v < 8; ++v) {
    float2 f = *(const float2*)(row + koff_b(v, half));
    r[2 * v] = f2bf(f.x);
    r[2 * v + 1] = f2bf(f.y);
  }
  return r;
}
DEV v16bf afrag_bf16(const unsigned short* A, long lda, int m0, int k0) {
  int lane = threadIdx.x & 31, half = lane >> 4;
  const unsigned* row = (const unsigned*)(A + (long)(m0 + (lane & 15)) * lda + k0);
  v8u u;
#pragma unroll
  for (int v = 0; v < 8; ++v) u[v] = row[koff_a(v, half) >> 1];
  return __builtin_bit_cast(v16bf, u);
}
DEV v16bf afrag_bf16_g(const unsigned short* A, long lda, int m0, int k0, int M) {
  int lane = threadIdx.x & 31, half = lane >> 4;
  int m = m0 + (lane & 15);
  v8u u = {0u, 0u, 0u, 0u, 0u, 0u, 0u, 0u};
  if (m < M) {
    const unsigned* row = (const unsigned*)(A + (long)m * lda + k0);
#pragma unroll
    for (int v = 0; v < 8; ++v) u[v] = row[koff_a(v, half) >> 1];
  }
  return __builtin_bit_cast(v16bf, u);
}
// B fragment from packed bf16 K-pairs P[(k>>1)*ldn + n]: one dword per VGPR
DEV v16bf bfrag_pk(const unsigned* P, long ldn, int n0, int k0) {
  int lane = threadIdx.x & 31, half = lane >> 4;
  const unsigned* col = P + n0 + (lane & 15);
  v8u u;
#pragma unroll
  for (int v = 0; v < 8; ++v)
    u[v] = col[(long)((k0 + koff_b(v, half)) >> 1) * ldn];
  return __builtin_bit_cast(v16bf, u);
}

DEV float wave_max(float x) {
#pragma unroll
  for (int o = 16; o > 0; o >>= 1) x = fmaxf(x, __shfl_xor(x, o, 32));
  return x;
}
DEV float wave_sum(float x) {
#pragma unroll
  for (int o = 16; o > 0; o >>= 1) x += __shfl_xor(x, o, 32);
  return x;
}
DEV float softplus(float x) { return log1pf(expf(x)); }

// ======================= constants =======================
#define NRES 512
#define NSURF 2048
#define NH 12
#define OCAT 1152
#define INF_ 100000.0f
#define SQRT_INV48 0.14433756729740643f     // sqrt(1/(3*C_H))
#define SQRT_INV54 0.13608276348795434f     // sqrt(1/(3*PQ*4.5))
#define SQRT13 0.5773502691896258f          // sqrt(1/3)

// ======================= weight pre-packing =======================
__global__ void k_packW(const float* W, unsigned* Wp, int K, int Ksrc, int N) {
  int idx = blockIdx.x * blockDim.x + threadIdx.x;
  if (idx >= (K >> 1) * N) return;
  int kp = idx / N, n = idx % N;
  float lo = (2 * kp     < Ksrc) ? W[(long)(2 * kp) * N + n] : 0.f;
  float hi = (2 * kp + 1 < Ksrc) ? W[(long)(2 * kp + 1) * N + n] : 0.f;
  Wp[idx] = pack2bf(lo, hi);
}

// ======================= generic bf16-WMMA GEMM (padded shapes) =======================
// one wave owns a 16-row strip and up to 4 N-tiles: A fragment shared by 4 WMMAs
template <bool ABF16>
__global__ void k_gemm(const void* Ap, long lda, const unsigned* Wp,
                       const float* bias, float* out, long ldo,
                       int M, int N, int K) {
  int wave = threadIdx.x >> 5, lane = threadIdx.x & 31;
  int ntn = N >> 4;
  int ngrp = (ntn + 3) >> 2;
  int idx = blockIdx.x * (blockDim.x >> 5) + wave;
  if (idx >= (M >> 4) * ngrp) return;
  int tm = (idx / ngrp) * 16;
  int g = idx % ngrp;
  int cnt = min(4, ntn - g * 4);              // wave-uniform
  v8f acc[4];
#pragma unroll
  for (int t = 0; t < 4; ++t) acc[t] = (v8f){0.f, 0.f, 0.f, 0.f, 0.f, 0.f, 0.f, 0.f};
  for (int k0 = 0; k0 < K; k0 += 32) {
    v16bf a;
    if (ABF16) a = afrag_bf16((const unsigned short*)Ap, lda, tm, k0);
    else       a = afrag_f32((const float*)Ap, lda, tm, k0);
#pragma unroll
    for (int t = 0; t < 4; ++t)
      if (t < cnt)
        acc[t] = wmma_bf16(a, bfrag_pk(Wp, N, g * 64 + t * 16, k0), acc[t]);
  }
  int half = lane >> 4, nc = lane & 15;
#pragma unroll
  for (int t = 0; t < 4; ++t) {
    if (t < cnt) {
      int col = g * 64 + t * 16 + nc;
      float bv = bias ? bias[col] : 0.0f;
#pragma unroll
      for (int r = 0; r < 8; ++r)
        out[(long)(tm + r + 8 * half) * ldo + col] = acc[t][r] + bv;
    }
  }
}

// ======================= z-projection (the 134MB stream) =======================
__global__ void k_pack_zw(const float* Wb, const float* bb, const float* Wdz,
                          const float* bdz, unsigned* Wzp, float* bz) {
  for (int idx = threadIdx.x; idx < 64 * 48; idx += blockDim.x) {
    int kp = idx / 48, c = idx % 48;
    int k0 = 2 * kp, k1 = 2 * kp + 1;
    float lo = (c < 12) ? Wb[k0 * 12 + c] : ((c < 44) ? Wdz[k0 * 32 + (c - 12)] : 0.f);
    float hi = (c < 12) ? Wb[k1 * 12 + c] : ((c < 44) ? Wdz[k1 * 32 + (c - 12)] : 0.f);
    Wzp[idx] = pack2bf(lo, hi);
  }
  for (int c = threadIdx.x; c < 48; c += blockDim.x)
    bz[c] = (c < 12) ? bb[c] : ((c < 44) ? bdz[c - 12] : 0.f);
}

DEV void zproj_store(v8f acc, int tn, long tm, float* bbias, unsigned* pz16,
                     const float* bz) {
  int lane = threadIdx.x & 31, half = lane >> 4;
  int col = tn + (lane & 15);
  float bv = bz[col];
  if (col < 12) {
#pragma unroll
    for (int r = 0; r < 8; ++r)
      bbias[(long)col * NRES * NRES + tm + r + 8 * half] = acc[r] + bv;
  } else if (col < 44) {
#pragma unroll
    for (int r = 0; r < 8; r += 2) {
      long prow = (tm + r + 8 * half) >> 1;   // rows r,r+1 are consecutive j's
      pz16[prow * 32 + (col - 12)] = pack2bf(acc[r] + bv, acc[r + 1] + bv);
    }
  }
}

__global__ void k_zproj(const float* z, const unsigned* Wzp, const float* bz,
                        float* bbias, unsigned* pz16) {
  int wave = threadIdx.x >> 5, lane = threadIdx.x & 31;
  long tm = ((long)blockIdx.x * 8 + wave) * 16;   // 16384 strips
  __builtin_prefetch(z + (tm + 16) * 128 + (lane << 2), 0, 1);  // global_prefetch_b8
  v8f acc0 = {0.f, 0.f, 0.f, 0.f, 0.f, 0.f, 0.f, 0.f};
  v8f acc1 = acc0, acc2 = acc0;
#pragma unroll
  for (int k0 = 0; k0 < 128; k0 += 32) {
    v16bf a = afrag_f32(z, 128, (int)tm, k0);
    acc0 = wmma_bf16(a, bfrag_pk(Wzp, 48, 0, k0), acc0);
    acc1 = wmma_bf16(a, bfrag_pk(Wzp, 48, 16, k0), acc1);
    acc2 = wmma_bf16(a, bfrag_pk(Wzp, 48, 32, k0), acc2);
  }
  zproj_store(acc0, 0, tm, bbias, pz16, bz);
  zproj_store(acc1, 16, tm, bbias, pz16, bz);
  zproj_store(acc2, 32, tm, bbias, pz16, bz);
}

// ======================= point transforms + operand packing =======================
__global__ void k_points(const float* qg, const float* kvg, const float* qpr,
                         const float* kvpr, const float* sqg, const float* rots,
                         const float* trans, const float* head_w, const float* sb_w,
                         float* Qcat, float* Kcat, float* Vcat, float* rq,
                         float* rk, float* Qs, float* rs) {
  int i = blockIdx.x, tid = threadIdx.x;
  __shared__ float R[9], T[3], accq[12], acck[12];
  if (tid < 9) R[tid] = rots[i * 9 + tid];
  if (tid < 3) T[tid] = trans[i * 3 + tid];
  if (tid < 12) { accq[tid] = 0.f; acck[tid] = 0.f; }
  __syncthreads();
  for (int p = tid; p < 48; p += blockDim.x) {      // 48 query points
    float v0 = qpr[i * 144 + p], v1 = qpr[i * 144 + 48 + p], v2 = qpr[i * 144 + 96 + p];
    float g0 = R[0] * v0 + R[1] * v1 + R[2] * v2 + T[0];
    float g1 = R[3] * v0 + R[4] * v1 + R[5] * v2 + T[1];
    float g2 = R[6] * v0 + R[7] * v1 + R[8] * v2 + T[2];
    int h = p >> 2, pp = p & 3;
    float wpt2 = softplus(head_w[h]) * SQRT_INV54;  // 2*(0.5*hw*sqrt(1/54))
    long b = ((long)h * NRES + i) * 32 + 16 + pp * 3;
    Qcat[b] = g0 * wpt2; Qcat[b + 1] = g1 * wpt2; Qcat[b + 2] = g2 * wpt2;
    atomicAdd(&accq[h], g0 * g0 + g1 * g1 + g2 * g2);
  }
  for (int p = tid; p < 144; p += blockDim.x) {     // 144 key/value points
    float v0 = kvpr[i * 432 + p], v1 = kvpr[i * 432 + 144 + p], v2 = kvpr[i * 432 + 288 + p];
    float g0 = R[0] * v0 + R[1] * v1 + R[2] * v2 + T[0];
    float g1 = R[3] * v0 + R[4] * v1 + R[5] * v2 + T[1];
    float g2 = R[6] * v0 + R[7] * v1 + R[8] * v2 + T[2];
    int h = p / 12, pj = p % 12;
    if (pj < 4) {
      long b = ((long)h * NRES + i) * 32 + 16 + pj * 3;
      Kcat[b] = g0; Kcat[b + 1] = g1; Kcat[b + 2] = g2;
      atomicAdd(&acck[h], g0 * g0 + g1 * g1 + g2 * g2);
    } else {
      long b = ((long)h * NRES + i) * 48 + 16 + (pj - 4) * 3;
      Vcat[b] = g0; Vcat[b + 1] = g1; Vcat[b + 2] = g2;
    }
  }
  float sbw = softplus(sb_w[0]);
  for (int idx = tid; idx < NH * 32; idx += blockDim.x) {
    int h = idx >> 5, c = idx & 31;
    long b = ((long)h * NRES + i) * 32 + c;
    if (c < 16) {
      Qcat[b] = qg[i * 192 + h * 16 + c] * SQRT_INV48;
      Kcat[b] = kvg[i * 384 + h * 16 + c];
      Qs[b] = sqg[i * 192 + h * 16 + c] * 0.25f;
    } else {
      if (c >= 28) { Qcat[b] = 0.f; Kcat[b] = 0.f; }
      Qs[b] = (c < 19) ? T[c - 16] * sbw : 0.f;
    }
  }
  for (int idx = tid; idx < NH * 48; idx += blockDim.x) {   // Vcat c<16 and pad 40..47
    int h = idx / 48, c = idx % 48;
    long b = ((long)h * NRES + i) * 48 + c;
    if (c < 16) Vcat[b] = kvg[i * 384 + 192 + h * 16 + c];
    else if (c >= 40) Vcat[b] = 0.f;
  }
  __syncthreads();
  if (tid < 12) { rq[tid * NRES + i] = accq[tid]; rk[tid * NRES + i] = acck[tid]; }
  if (tid == 0) rs[i] = -0.5f * sbw * (T[0] * T[0] + T[1] * T[1] + T[2] * T[2]);
}

__global__ void k_surf_pack(const float* skg, const float* spts, const float* sb_w,
                            float* Ks, float* rms) {
  int m = blockIdx.x, tid = threadIdx.x;
  float p0 = spts[m * 3], p1 = spts[m * 3 + 1], p2 = spts[m * 3 + 2];
  for (int idx = tid; idx < NH * 32; idx += blockDim.x) {
    int h = idx >> 5, c = idx & 31;
    float v = 0.f;
    if (c < 16) v = skg[m * 192 + h * 16 + c];
    else if (c == 16) v = p0; else if (c == 17) v = p1; else if (c == 18) v = p2;
    Ks[((long)h * NSURF + m) * 32 + c] = v;
  }
  if (tid == 0) rms[m] = -0.5f * softplus(sb_w[0]) * (p0 * p0 + p1 * p1 + p2 * p2);
}

__global__ void k_pad_sf(const float* sfin, float* sfp) {
  int idx = blockIdx.x * blockDim.x + threadIdx.x;
  if (idx >= NSURF * 32) return;
  int m = idx >> 5, c = idx & 31;
  sfp[idx] = (c < 16) ? sfin[m * 16 + c] : 0.f;
}

// ======================= IPA logits + softmax =======================
__global__ void k_attn(const float* Qcat, const float* Kcat, const float* bbias,
                       const float* rq, const float* rk, const float* mask,
                       const float* head_w, unsigned short* a16) {
  int h = blockIdx.y, tm = blockIdx.x * 16;
  int wave = threadIdx.x >> 5, lane = threadIdx.x & 31;
  int half = lane >> 4, nc = lane & 15;
  __shared__ float lds[16][NRES];
  const float* Qh = Qcat + (long)h * NRES * 32;
  const float* Kh = Kcat + (long)h * NRES * 32;
  const float* Bh = bbias + (long)h * NRES * NRES;
  v16bf a = afrag_f32(Qh, 32, tm, 0);
  float wpt = 0.5f * SQRT_INV54 * softplus(head_w[h]);
#pragma unroll
  for (int tt = 0; tt < 4; ++tt) {
    int tn = wave * 64 + tt * 16;
    v16bf b = bfrag_nmaj(Kh, 32, tn, 0);     // Kcat stored [j][c] (N-major)
    v8f acc = {0.f, 0.f, 0.f, 0.f, 0.f, 0.f, 0.f, 0.f};
    acc = wmma_bf16(a, b, acc);
    int col = tn + nc;
    float cb = -wpt * rk[h * NRES + col];
    float cm = mask[col];
#pragma unroll
    for (int r = 0; r < 8; ++r) {
      int lr = r + 8 * half, row = tm + lr;
      lds[lr][col] = acc[r] + SQRT13 * Bh[(long)row * NRES + col]
                   - wpt * rq[h * NRES + row] + cb
                   + INF_ * (mask[row] * cm - 1.0f);
    }
  }
  __syncthreads();
  for (int rr = wave * 2; rr < wave * 2 + 2; ++rr) {
    float vals[16], mx = -3.0e38f;
#pragma unroll
    for (int t = 0; t < 16; ++t) { vals[t] = lds[rr][lane + t * 32]; mx = fmaxf(mx, vals[t]); }
    mx = wave_max(mx);
    float sm = 0.f;
#pragma unroll
    for (int t = 0; t < 16; ++t) { vals[t] = __expf(vals[t] - mx); sm += vals[t]; }
    float inv = 1.0f / wave_sum(sm);
    long base = ((long)h * NRES + (tm + rr)) * NRES;
#pragma unroll
    for (int t = 0; t < 16; ++t) a16[base + lane + t * 32] = bf2bits(f2bf(vals[t] * inv));
  }
}

// o = A@[v|vp] per head: one wave owns a 16-row strip; 3 N-tiles share A fragments
__global__ void k_av(const unsigned short* a16, const float* Vcat, float* ocat,
                     float* optsum) {
  int wave = threadIdx.x >> 5, lane = threadIdx.x & 31;
  int strip = blockIdx.x * 4 + wave;           // 12*32 = 384 strips
  int h = strip / 32, tm = (strip % 32) * 16;
  const unsigned short* Ah = a16 + (long)h * NRES * NRES;
  const float* Vh = Vcat + (long)h * NRES * 48;
  v8f acc0 = {0.f, 0.f, 0.f, 0.f, 0.f, 0.f, 0.f, 0.f};
  v8f acc1 = acc0, acc2 = acc0;
  for (int k0 = 0; k0 < NRES; k0 += 32) {
    v16bf a = afrag_bf16(Ah, NRES, tm, k0);
    acc0 = wmma_bf16(a, bfrag_nmaj(Vh, 48, 0, k0), acc0);   // Vcat is [j][c]
    acc1 = wmma_bf16(a, bfrag_nmaj(Vh, 48, 16, k0), acc1);
    acc2 = wmma_bf16(a, bfrag_nmaj(Vh, 48, 32, k0), acc2);
  }
  int half = lane >> 4, nc = lane & 15;
  v8f accs[3] = {acc0, acc1, acc2};
#pragma unroll
  for (int t = 0; t < 3; ++t) {
    int col = t * 16 + nc;
#pragma unroll
    for (int r = 0; r < 8; ++r) {
      int row = tm + r + 8 * half;
      if (col < 16)      ocat[(long)row * OCAT + h * 16 + col] = accs[t][r];
      else if (col < 40) optsum[((long)row * NH + h) * 24 + (col - 16)] = accs[t][r];
    }
  }
}

// o_pair[i] = A_i(12x512) @ pair_z_i(512x32); 2 N-tiles share A fragments
__global__ void k_opair(const unsigned short* a16, const unsigned* pz16, float* ocat) {
  int wave = threadIdx.x >> 5, lane = threadIdx.x & 31;
  int i = blockIdx.x * 4 + wave;               // 512 strips
  const unsigned* Pz = pz16 + (long)i * 256 * 32;
  v8f acc0 = {0.f, 0.f, 0.f, 0.f, 0.f, 0.f, 0.f, 0.f};
  v8f acc1 = acc0;
  for (int k0 = 0; k0 < NRES; k0 += 32) {
    v16bf a = afrag_bf16_g(a16 + (long)i * NRES, (long)NRES * NRES, 0, k0, NH);
    acc0 = wmma_bf16(a, bfrag_pk(Pz, 32, 0, k0), acc0);
    acc1 = wmma_bf16(a, bfrag_pk(Pz, 32, 16, k0), acc1);
  }
  int half = lane >> 4, nc = lane & 15;
  v8f accs[2] = {acc0, acc1};
#pragma unroll
  for (int t = 0; t < 2; ++t) {
    int col = t * 16 + nc;
#pragma unroll
    for (int r = 0; r < 8; ++r) {
      int hrow = r + 8 * half;
      if (hrow < NH) ocat[(long)i * OCAT + 576 + hrow * 32 + col] = accs[t][r];
    }
  }
}

// ======================= surface attention =======================
__global__ void k_slogits(const float* Qs, const float* Ks, const float* rs,
                          const float* rms, const float* mask, const float* smask,
                          float* slog) {
  int h = blockIdx.z, tm = blockIdx.y * 16, mch = blockIdx.x;
  int wave = threadIdx.x >> 5, lane = threadIdx.x & 31;
  int half = lane >> 4, nc = lane & 15;
  const float* Qh = Qs + (long)h * NRES * 32;
  const float* Kh = Ks + (long)h * NSURF * 32;
  v16bf a = afrag_f32(Qh, 32, tm, 0);
#pragma unroll
  for (int tt = 0; tt < 4; ++tt) {
    int tn = mch * 512 + wave * 64 + tt * 16;
    v16bf b = bfrag_nmaj(Kh, 32, tn, 0);     // Ks stored [m][c] (N-major)
    v8f acc = {0.f, 0.f, 0.f, 0.f, 0.f, 0.f, 0.f, 0.f};
    acc = wmma_bf16(a, b, acc);
    int col = tn + nc;
    float cb = rms[col], cm = smask[col];
#pragma unroll
    for (int r = 0; r < 8; ++r) {
      int row = tm + r + 8 * half;
      slog[((long)h * NRES + row) * NSURF + col] =
          acc[r] + rs[row] + cb + INF_ * (mask[row] * cm - 1.0f);
    }
  }
}

__global__ void k_softmax_m(const float* slog, unsigned short* as16) {
  int wave = threadIdx.x >> 5, lane = threadIdx.x & 31;
  long row = (long)blockIdx.x * 4 + wave;      // 6144 rows
  const float* p = slog + row * NSURF;
  float vals[64], mx = -3.0e38f;
#pragma unroll
  for (int t = 0; t < 64; ++t) { vals[t] = p[lane + t * 32]; mx = fmaxf(mx, vals[t]); }
  mx = wave_max(mx);
  float sm = 0.f;
#pragma unroll
  for (int t = 0; t < 64; ++t) { vals[t] = __expf(vals[t] - mx); sm += vals[t]; }
  float inv = 1.0f / wave_sum(sm);
  unsigned short* o = as16 + row * NSURF;
#pragma unroll
  for (int t = 0; t < 64; ++t) o[lane + t * 32] = bf2bits(f2bf(vals[t] * inv));
}

__global__ void k_softmax_n(const float* slog, unsigned short* at16) {
  int wave = threadIdx.x >> 5, lane = threadIdx.x & 31;
  int m = blockIdx.x * 4 + wave;               // 2048
  float acc[16];
#pragma unroll
  for (int t = 0; t < 16; ++t) acc[t] = 0.f;
  for (int h = 0; h < NH; ++h) {
    const float* p = slog + (long)h * NRES * NSURF + m;
    float vals[16], mx = -3.0e38f;
#pragma unroll
    for (int t = 0; t < 16; ++t) {
      vals[t] = p[(long)(lane + t * 32) * NSURF];
      mx = fmaxf(mx, vals[t]);
    }
    mx = wave_max(mx);
    float sm = 0.f;
#pragma unroll
    for (int t = 0; t < 16; ++t) { vals[t] = __expf(vals[t] - mx); sm += vals[t]; }
    float inv = 1.0f / wave_sum(sm);
#pragma unroll
    for (int t = 0; t < 16; ++t) acc[t] += vals[t] * inv;
  }
#pragma unroll
  for (int t = 0; t < 16; ++t)
    at16[(long)m * NRES + lane + t * 32] = bf2bits(f2bf(acc[t] * (1.0f / 12.0f)));
}

// o_surf = a_s @ sv per head: M=512, K=2048, N=16 (sv pre-packed K-pairs, ldn=192)
__global__ void k_osurf(const unsigned short* as16, const unsigned* svp, float* ocat) {
  int wave = threadIdx.x >> 5, lane = threadIdx.x & 31;
  int tile = blockIdx.x * 4 + wave;            // 12*32 = 384 tiles
  int h = tile / 32, tm = (tile % 32) * 16;
  const unsigned short* Ah = as16 + (long)h * NRES * NSURF;
  v8f acc = {0.f, 0.f, 0.f, 0.f, 0.f, 0.f, 0.f, 0.f};
  for (int k0 = 0; k0 < NSURF; k0 += 32) {
    v16bf a = afrag_bf16(Ah, NSURF, tm, k0);
    v16bf b = bfrag_pk(svp + h * 16, 192, 0, k0);
    acc = wmma_bf16(a, b, acc);
  }
  int half = lane >> 4, col = lane & 15;
#pragma unroll
  for (int r = 0; r < 8; ++r)
    ocat[(long)(tm + r + 8 * half) * OCAT + 960 + h * 16 + col] = acc[r];
}

// inverse-rotate o_pt, norms -> o_cat[192..575]
__global__ void k_finalize(const float* optsum, const float* rots, const float* trans,
                           float* ocat) {
  int i = blockIdx.x, tid = threadIdx.x;
  __shared__ float R[9], T[3];
  if (tid < 9) R[tid] = rots[i * 9 + tid];
  if (tid < 3) T[tid] = trans[i * 3 + tid];
  __syncthreads();
  for (int idx = tid; idx < 96; idx += blockDim.x) {
    int h = idx >> 3, p = idx & 7;
    const float* o = optsum + ((long)i * NH + h) * 24 + p * 3;
    float v0 = o[0] - T[0], v1 = o[1] - T[1], v2 = o[2] - T[2];
    float l0 = R[0] * v0 + R[3] * v1 + R[6] * v2;   // R^T
    float l1 = R[1] * v0 + R[4] * v1 + R[7] * v2;
    float l2 = R[2] * v0 + R[5] * v1 + R[8] * v2;
    float nrm = sqrtf(l0 * l0 + l1 * l1 + l2 * l2 + 1e-8f);
    long b = (long)i * OCAT;
    int hp = h * 8 + p;
    ocat[b + 192 + hp] = l0;
    ocat[b + 288 + hp] = l1;
    ocat[b + 384 + hp] = l2;
    ocat[b + 480 + hp] = nrm;
  }
}

__global__ void k_pack_scat(const float* sfeat, const float* sagg, float* scat) {
  long idx = (long)blockIdx.x * blockDim.x + threadIdx.x;
  if (idx >= (long)NSURF * 768) return;
  long m = idx / 768; int c = (int)(idx % 768);
  scat[idx] = (c < 384) ? sfeat[m * 384 + c] : sagg[m * 384 + (c - 384)];
}

// ======================= host launcher =======================
static void packW(const float* W, unsigned* Wp, int K, int Ksrc, int N, hipStream_t s) {
  int tot = (K / 2) * N;
  k_packW<<<(tot + 255) / 256, 256, 0, s>>>(W, Wp, K, Ksrc, N);
}
static void gemm_f32(const float* A, long lda, const unsigned* Wp,
                     const float* bias, float* out, long ldo, int M, int N, int K,
                     hipStream_t s) {
  int ngrp = (N / 16 + 3) / 4;
  int waves = (M / 16) * ngrp;
  k_gemm<false><<<(waves + 3) / 4, 128, 0, s>>>(A, lda, Wp, bias, out, ldo, M, N, K);
}

extern "C" void kernel_launch(void* const* d_in, const int* in_sizes, int n_in,
                              void* d_out, int out_size, void* d_ws, size_t ws_size,
                              hipStream_t stream) {
  const float* s_in  = (const float*)d_in[0];
  const float* z     = (const float*)d_in[1];
  const float* rots  = (const float*)d_in[2];
  const float* trans = (const float*)d_in[3];
  const float* mask  = (const float*)d_in[4];
  const float* spts  = (const float*)d_in[5];
  const float* sfin  = (const float*)d_in[6];
  const float* smask = (const float*)d_in[7];
  const float *Wq = (const float*)d_in[8],  *bq = (const float*)d_in[9];
  const float *Wkv = (const float*)d_in[10], *bkv = (const float*)d_in[11];
  const float *Wqp = (const float*)d_in[12], *bqp = (const float*)d_in[13];
  const float *Wkvp = (const float*)d_in[14], *bkvp = (const float*)d_in[15];
  const float *Wse = (const float*)d_in[16], *bse = (const float*)d_in[17];
  const float *Wsq = (const float*)d_in[18], *bsq = (const float*)d_in[19];
  const float *Wsk = (const float*)d_in[20], *bsk = (const float*)d_in[21];
  const float *Wsv = (const float*)d_in[22], *bsv = (const float*)d_in[23];
  const float *Wb = (const float*)d_in[24],  *bb = (const float*)d_in[25];
  const float *Wdz = (const float*)d_in[26], *bdz = (const float*)d_in[27];
  const float *hw = (const float*)d_in[28],  *sbw = (const float*)d_in[29];
  const float *Wout = (const float*)d_in[30], *bout = (const float*)d_in[31];
  const float *Wso = (const float*)d_in[32],  *bso = (const float*)d_in[33];

  float* w = (float*)d_ws;
  float* q    = w; w += 512 * 192;
  float* kv   = w; w += 512 * 384;
  float* qpr  = w; w += 512 * 144;
  float* kvpr = w; w += 512 * 432;
  float* sqb  = w; w += 512 * 192;
  float* sfp  = w; w += 2048L * 32;
  float* sfeat= w; w += 2048L * 384;
  float* sk   = w; w += 2048L * 192;
  float* sv   = w; w += 2048L * 192;
  float* Qcat = w; w += 12L * 512 * 32;
  float* Kcat = w; w += 12L * 512 * 32;
  float* Vcat = w; w += 12L * 512 * 48;
  float* rq   = w; w += 12 * 512;
  float* rk   = w; w += 12 * 512;
  float* Qs   = w; w += 12L * 512 * 32;
  float* Ks   = w; w += 12L * 2048 * 32;
  float* rs   = w; w += 512;
  float* rms  = w; w += 2048;
  float* bbias= w; w += 12L * 512 * 512;
  unsigned* pz16 = (unsigned*)w; w += 512L * 256 * 32;          // packed bf16 pairs
  unsigned short* a16 = (unsigned short*)w; w += (12L * 512 * 512) / 2;
  float* optsum = w; w += 512L * 12 * 24;
  float* slog = w; w += 12L * 512 * 2048;
  unsigned short* as16 = (unsigned short*)w; w += (12L * 512 * 2048) / 2;
  unsigned short* at16 = (unsigned short*)w; w += (2048L * 512) / 2;
  float* sagg = w; w += 2048L * 384;
  float* ocat = w; w += 512L * 1152;
  float* scat = w; w += 2048L * 768;
  // packed bf16 weights (dwords)
  unsigned* pWq   = (unsigned*)w; w += 192 * 192;
  unsigned* pWkv  = (unsigned*)w; w += 192 * 384;
  unsigned* pWqp  = (unsigned*)w; w += 192 * 144;
  unsigned* pWkvp = (unsigned*)w; w += 192 * 432;
  unsigned* pWsq  = (unsigned*)w; w += 192 * 192;
  unsigned* pWse  = (unsigned*)w; w += 16 * 384;
  unsigned* pWsk  = (unsigned*)w; w += 192 * 192;
  unsigned* pWsv  = (unsigned*)w; w += 192 * 192;
  unsigned* pWout = (unsigned*)w; w += 576 * 384;
  unsigned* pWso  = (unsigned*)w; w += 384 * 16;
  unsigned* psin  = (unsigned*)w; w += 256 * 384;
  unsigned* psv   = (unsigned*)w; w += 1024 * 192;
  unsigned* Wzp   = (unsigned*)w; w += 64 * 48;
  float* bz = w; w += 48;

  float* out_s    = (float*)d_out;
  float* out_surf = out_s + 512 * 384;

  // ---- stage 0: pack all K-major B operands to bf16 K-pairs ----
  k_pack_zw<<<1, 128, 0, stream>>>(Wb, bb, Wdz, bdz, Wzp, bz);
  packW(Wq, pWq, 384, 384, 192, stream);
  packW(Wkv, pWkv, 384, 384, 384, stream);
  packW(Wqp, pWqp, 384, 384, 144, stream);
  packW(Wkvp, pWkvp, 384, 384, 432, stream);
  packW(Wsq, pWsq, 384, 384, 192, stream);
  packW(Wse, pWse, 32, 16, 384, stream);
  packW(Wsk, pWsk, 384, 384, 192, stream);
  packW(Wsv, pWsv, 384, 384, 192, stream);
  packW(Wout, pWout, 1152, 1152, 384, stream);
  packW(Wso, pWso, 768, 768, 16, stream);
  packW(s_in, psin, 512, 512, 384, stream);
  k_pad_sf<<<(NSURF * 32 + 255) / 256, 256, 0, stream>>>(sfin, sfp);

  // ---- stage 1: projections (bf16 WMMA GEMMs, all shapes padded) ----
  gemm_f32(s_in, 384, pWq, bq, q, 192, 512, 192, 384, stream);
  gemm_f32(s_in, 384, pWkv, bkv, kv, 384, 512, 384, 384, stream);
  gemm_f32(s_in, 384, pWqp, bqp, qpr, 144, 512, 144, 384, stream);
  gemm_f32(s_in, 384, pWkvp, bkvp, kvpr, 432, 512, 432, 384, stream);
  gemm_f32(s_in, 384, pWsq, bsq, sqb, 192, 512, 192, 384, stream);
  gemm_f32(sfp, 32, pWse, bse, sfeat, 384, 2048, 384, 32, stream);
  gemm_f32(sfeat, 384, pWsk, bsk, sk, 192, 2048, 192, 384, stream);
  gemm_f32(sfeat, 384, pWsv, bsv, sv, 192, 2048, 192, 384, stream);
  packW(sv, psv, 2048, 2048, 192, stream);   // sv as packed B for o_surf

  // ---- stage 2: operand packing (rigid transforms, scale folds) ----
  k_points<<<512, 256, 0, stream>>>(q, kv, qpr, kvpr, sqb, rots, trans, hw, sbw,
                                    Qcat, Kcat, Vcat, rq, rk, Qs, rs);
  k_surf_pack<<<2048, 256, 0, stream>>>(sk, spts, sbw, Ks, rms);

  // ---- stage 3: the big z stream (fused b_bias + pair_z) ----
  k_zproj<<<2048, 256, 0, stream>>>(z, Wzp, bz, bbias, pz16);

  // ---- stage 4: IPA attention ----
  k_attn<<<dim3(32, 12), 256, 0, stream>>>(Qcat, Kcat, bbias, rq, rk, mask, hw, a16);
  k_av<<<96, 128, 0, stream>>>(a16, Vcat, ocat, optsum);
  k_opair<<<128, 128, 0, stream>>>(a16, pz16, ocat);
  k_finalize<<<512, 128, 0, stream>>>(optsum, rots, trans, ocat);

  // ---- stage 5: surface attention ----
  k_slogits<<<dim3(4, 32, 12), 256, 0, stream>>>(Qs, Ks, rs, rms, mask, smask, slog);
  k_softmax_m<<<1536, 128, 0, stream>>>(slog, as16);
  k_softmax_n<<<512, 128, 0, stream>>>(slog, at16);
  k_osurf<<<96, 128, 0, stream>>>(as16, psv, ocat);
  {
    int ngrp = 6;                              // N=384 -> 24 tiles -> 6 groups
    int waves = (2048 / 16) * ngrp;
    k_gemm<true><<<(waves + 3) / 4, 128, 0, stream>>>(at16, 512L, psin, nullptr,
                                                      sagg, 384, 2048, 384, 512);
  }

  // ---- stage 6: output projections ----
  gemm_f32(ocat, 1152, pWout, bout, out_s, 384, 512, 384, 1152, stream);
  k_pack_scat<<<(int)((2048L * 768 + 255) / 256), 256, 0, stream>>>(sfeat, sagg, scat);
  gemm_f32(scat, 768, pWso, bso, out_surf, 16, 2048, 16, 768, stream);
}
